// AccumBiRNNClass_89223650607930
// MI455X (gfx1250) — matrix operand order repaired and checked by reading, not compiled, not run on hardware
//
#include <hip/hip_runtime.h>

// ---------------------------------------------------------------------------
// Bidirectional RNN for MI455X (gfx1250), wave32 + WMMA bf16.
//
//  B=64, T=1024, D=512.  out[b][t][0:512]=fwd h_t, [512:1024]=bwd h_t (f32).
//
//  Phase 1 (xproj): xp = x @ Wih^T + (bih+bhh), written straight into d_out.
//  Phase 2 (rnn):   h_t = tanh(xp_t + h_{t-1} @ Whh^T), persistent kernel,
//                   Whh resident in VGPRs (bf16), split grid barrier per step
//                   with next-step xp loads hidden inside the barrier window.
//
//  d_ws layout (bytes):
//    [0,        1MB)    Wih_f, Wih_b   bf16
//    [1MB,      2MB)    Whh_f, Whh_b   bf16
//    [2MB,      2MB+4K) bias (bih+bhh) f32, 2x512
//    [..,  +256KB)      h double buffer: [dir][parity][64][512] bf16
//    [..,  +16B)        grid barrier {count, generation}
//  total ~2.4 MB.
// ---------------------------------------------------------------------------

typedef unsigned short u16;
typedef __bf16    v16bf    __attribute__((ext_vector_type(16)));
typedef float     v8f      __attribute__((ext_vector_type(8)));
typedef unsigned  uint32x4 __attribute__((ext_vector_type(4)));
typedef unsigned  uint32x2 __attribute__((ext_vector_type(2)));

#define NW (512 * 512)

union ABfrag { uint32x4 q[2]; v16bf v; };

__device__ __forceinline__ u16 f2bf(float f) {  // round-to-nearest-even
  unsigned u = __float_as_uint(f);
  u += 0x7FFFu + ((u >> 16) & 1u);
  return (u16)(u >> 16);
}
__device__ __forceinline__ unsigned pack2(float a, float b) {
  return (unsigned)f2bf(a) | ((unsigned)f2bf(b) << 16);
}
__device__ __forceinline__ v8f wmma_bf16(v16bf a, v16bf b, v8f c) {
  // D = A(16x32 bf16) x B(32x16 bf16) + C(f32)
  return __builtin_amdgcn_wmma_f32_16x16x32_bf16(false, a, false, b,
                                                 (short)0, c, false, false);
}
__device__ __forceinline__ float fast_tanh(float x) {
  // tanh(x) = 1 - 2/(exp(2x)+1); exp -> v_exp_f32 (TRANS, co-executes w/ VALU)
  float e = __expf(2.0f * x);
  return 1.0f - 2.0f * __builtin_amdgcn_rcpf(e + 1.0f);
}

// ---------------------------------------------------------------------------
// prep: f32 -> bf16 weights, bias fold, barrier re-arm (deterministic/launch)
// ---------------------------------------------------------------------------
__global__ void prep_kernel(const float* __restrict__ wif, const float* __restrict__ whf,
                            const float* __restrict__ bif, const float* __restrict__ bhf,
                            const float* __restrict__ wib, const float* __restrict__ whb,
                            const float* __restrict__ bib, const float* __restrict__ bhb,
                            u16* __restrict__ Wih, u16* __restrict__ Whh,
                            float* __restrict__ bias, unsigned* __restrict__ bar) {
  int tid = blockIdx.x * 256 + threadIdx.x;
  if (tid < 2 * NW) {
    const float* s = (tid < NW) ? wif : wib;
    Wih[tid] = f2bf(s[tid & (NW - 1)]);
  } else if (tid < 4 * NW) {
    int e = tid - 2 * NW;
    const float* s = (e < NW) ? whf : whb;
    Whh[e] = f2bf(s[e & (NW - 1)]);
  } else if (tid < 4 * NW + 1024) {
    int i = tid - 4 * NW;
    int n = i & 511;
    bias[i] = (i < 512) ? (bif[n] + bhf[n]) : (bib[n] + bhb[n]);
  } else if (tid < 4 * NW + 1024 + 2) {
    bar[tid - (4 * NW + 1024)] = 0u;
  }
}

// ---------------------------------------------------------------------------
// xproj: xp = x @ Wih^T + bias  ->  d_out[row][dir*512 + n]
// grid (1024 row-tiles, 2 dirs), 512 threads (16 waves).
// Wave w owns n-tiles {2w, 2w+1} x all 4 m-tiles; LDS double-buffered so one
// __syncthreads per K-chunk.
// ---------------------------------------------------------------------------
__global__ __launch_bounds__(512) void xproj_kernel(const float* __restrict__ x,
                                                    const u16* __restrict__ Wih,
                                                    const float* __restrict__ bias,
                                                    float* __restrict__ out) {
  __shared__ __align__(16) u16 lds[2][64 * 32];  // double-buffered x chunk, bf16
  const int dir = blockIdx.y;
  const int rowbase = blockIdx.x * 64;
  const int wave = threadIdx.x >> 5;
  const int lane = threadIdx.x & 31;
  const int l15 = lane & 15;
  const int hi = lane >> 4;
  const u16* W = Wih + (size_t)dir * NW;
  const int ntile0 = wave * 2;

  const int srow = threadIdx.x >> 3;        // staging: 4 elems/thread
  const int sk4 = (threadIdx.x & 7) << 2;
  const float* xrow = x + (size_t)(rowbase + srow) * 512 + sk4;

  const v8f zero = {0.f, 0.f, 0.f, 0.f, 0.f, 0.f, 0.f, 0.f};
  v8f acc[8];
#pragma unroll
  for (int i = 0; i < 8; ++i) acc[i] = zero;

  for (int c = 0; c < 16; ++c) {
    {  // stage x chunk (64 rows x 32 K) into buffer c&1
      float4 f = *(const float4*)(xrow + c * 32);
      uint32x2 p;
      p.x = pack2(f.x, f.y);
      p.y = pack2(f.z, f.w);
      *(uint32x2*)(&lds[c & 1][srow * 32 + sk4]) = p;
      if (c < 15) __builtin_prefetch(xrow + (c + 1) * 32, 0, 1);
    }
    __syncthreads();

    // B fragments: lane holds Wih[n][32c + 16*hi .. +15] (contiguous)
    ABfrag b0, b1;
    {
      const uint32x4* bp0 =
          (const uint32x4*)(W + (size_t)(ntile0 * 16 + l15) * 512 + c * 32 + 16 * hi);
      b0.q[0] = bp0[0];
      b0.q[1] = bp0[1];
      const uint32x4* bp1 =
          (const uint32x4*)(W + (size_t)((ntile0 + 1) * 16 + l15) * 512 + c * 32 + 16 * hi);
      b1.q[0] = bp1[0];
      b1.q[1] = bp1[1];
    }
#pragma unroll
    for (int m = 0; m < 4; ++m) {
      // A fragment: lane row (16m+l15), K run [16*hi, 16*hi+16) contiguous
      ABfrag a;
      const uint32x4* ap = (const uint32x4*)(&lds[c & 1][(m * 16 + l15) * 32 + 16 * hi]);
      a.q[0] = ap[0];
      a.q[1] = ap[1];
      acc[m * 2 + 0] = wmma_bf16(a.v, b0.v, acc[m * 2 + 0]);
      acc[m * 2 + 1] = wmma_bf16(a.v, b1.v, acc[m * 2 + 1]);
    }
  }

#pragma unroll
  for (int j = 0; j < 2; ++j) {
    int col = (ntile0 + j) * 16 + l15;
    float bv = bias[dir * 512 + col];
#pragma unroll
    for (int m = 0; m < 4; ++m) {
#pragma unroll
      for (int r = 0; r < 8; ++r) {
        int rowg = rowbase + m * 16 + r + 8 * hi;  // C layout: M = r + 8*hi
        out[(size_t)rowg * 1024 + dir * 512 + col] = acc[m * 2 + j][r] + bv;
      }
    }
  }
}

// ---------------------------------------------------------------------------
// rnn: persistent. 8 blocks (dir = blk>>2), 256 threads (8 waves).
// Wave owns one 16-col tile; Whh slice resident in 16 B-fragments (128 VGPRs).
// Split grid barrier: arrive -> (store h_t f32, load xp_{t+1}) -> wait, so the
// HBM latency of next-step xp is hidden inside the sync window.
// ---------------------------------------------------------------------------
__global__ __launch_bounds__(256) void rnn_kernel(const u16* __restrict__ Whh,
                                                  u16* __restrict__ hbuf,
                                                  float* __restrict__ out,
                                                  unsigned* __restrict__ bar) {
  const int dir = blockIdx.x >> 2;
  const int nblk = blockIdx.x & 3;
  const int wave = threadIdx.x >> 5;
  const int lane = threadIdx.x & 31;
  const int l15 = lane & 15;
  const int hi = lane >> 4;
  const int ntile = nblk * 8 + wave;      // 0..31
  const int col = ntile * 16 + l15;       // output column / Whh row
  const u16* W = Whh + (size_t)dir * NW;
  u16* hb = hbuf + (size_t)dir * 65536;   // [parity][64][512] bf16
  float* outc = out + dir * 512 + col;    // column base in d_out
  unsigned* cnt = bar;
  unsigned* gen = bar + 1;

  // Resident B fragments: Whh[col][32c + 16*hi .. +15] contiguous.
  ABfrag B[16];
#pragma unroll
  for (int c = 0; c < 16; ++c) {
    const uint32x4* bp = (const uint32x4*)(W + (size_t)col * 512 + c * 32 + 16 * hi);
    B[c].q[0] = bp[0];
    B[c].q[1] = bp[1];
  }

  // Preload xp for step 0.
  float xp[4][8];
  {
    const int t0 = (dir == 0) ? 0 : 1023;
#pragma unroll
    for (int m = 0; m < 4; ++m)
#pragma unroll
      for (int r = 0; r < 8; ++r) {
        int b = m * 16 + r + 8 * hi;
        xp[m][r] = outc[(size_t)(b * 1024 + t0) * 1024];
      }
  }

  const v8f zero = {0.f, 0.f, 0.f, 0.f, 0.f, 0.f, 0.f, 0.f};
  for (int i = 0; i < 1024; ++i) {
    const int t = (dir == 0) ? i : (1023 - i);
    const u16* hr = hb + (size_t)((i & 1) ^ 1) * 32768;
    u16* hw = hb + (size_t)(i & 1) * 32768;

    v8f acc[4] = {zero, zero, zero, zero};
    if (i != 0) {  // i==0: h_{-1} == 0, skip GEMM (hbuf never read cold)
#pragma unroll
      for (int c = 0; c < 16; ++c) {
#pragma unroll
        for (int m = 0; m < 4; ++m) {
          ABfrag a;
          const uint32x4* ap =
              (const uint32x4*)(hr + (size_t)(m * 16 + l15) * 512 + c * 32 + 16 * hi);
          a.q[0] = ap[0];
          a.q[1] = ap[1];  // +8 elements (contiguous K run)
          acc[m] = wmma_bf16(a.v, B[c].v, acc[m]);
        }
      }
    }

    // h_t = tanh(acc + xp), in registers.
#pragma unroll
    for (int m = 0; m < 4; ++m)
#pragma unroll
      for (int r = 0; r < 8; ++r) acc[m][r] = fast_tanh(acc[m][r] + xp[m][r]);

    if (i == 1023) {  // last step: no successor, just emit f32 output
#pragma unroll
      for (int m = 0; m < 4; ++m)
#pragma unroll
        for (int r = 0; r < 8; ++r) {
          int b = m * 16 + r + 8 * hi;
          outc[(size_t)(b * 1024 + t) * 1024] = acc[m][r];
        }
      break;
    }

    // Publish h_t (bf16) for all blocks.
#pragma unroll
    for (int m = 0; m < 4; ++m)
#pragma unroll
      for (int r = 0; r < 8; ++r) {
        int b = m * 16 + r + 8 * hi;
        hw[b * 512 + col] = f2bf(acc[m][r]);
      }

    // ---- barrier arrive ----
    __threadfence();
    __syncthreads();
    unsigned g = 0;
    bool last_arriver = false;
    if (threadIdx.x == 0) {
      g = __hip_atomic_load(gen, __ATOMIC_RELAXED, __HIP_MEMORY_SCOPE_AGENT);
      unsigned a = __hip_atomic_fetch_add(cnt, 1u, __ATOMIC_ACQ_REL, __HIP_MEMORY_SCOPE_AGENT);
      if (a == 7u) {
        __hip_atomic_store(cnt, 0u, __ATOMIC_RELAXED, __HIP_MEMORY_SCOPE_AGENT);
        __hip_atomic_fetch_add(gen, 1u, __ATOMIC_ACQ_REL, __HIP_MEMORY_SCOPE_AGENT);
        last_arriver = true;
      }
    }

    // ---- overlap window: private f32 h_t stores + next-step xp loads ----
    const int tn = (dir == 0) ? (i + 1) : (1023 - (i + 1));
#pragma unroll
    for (int m = 0; m < 4; ++m)
#pragma unroll
      for (int r = 0; r < 8; ++r) {
        int b = m * 16 + r + 8 * hi;
        outc[(size_t)(b * 1024 + t) * 1024] = acc[m][r];   // h_t (private slot)
        xp[m][r] = outc[(size_t)(b * 1024 + tn) * 1024];   // xp_{t+1} (private)
      }

    // ---- barrier wait ----
    if (threadIdx.x == 0 && !last_arriver) {
      while (__hip_atomic_load(gen, __ATOMIC_ACQUIRE, __HIP_MEMORY_SCOPE_AGENT) == g)
        __builtin_amdgcn_s_sleep(1);
    }
    __syncthreads();
    __threadfence();
  }
}

// ---------------------------------------------------------------------------
extern "C" void kernel_launch(void* const* d_in, const int* in_sizes, int n_in,
                              void* d_out, int out_size, void* d_ws, size_t ws_size,
                              hipStream_t stream) {
  (void)in_sizes; (void)n_in; (void)out_size; (void)ws_size;
  const float* x     = (const float*)d_in[0];
  const float* Wih_f = (const float*)d_in[1];
  const float* Whh_f = (const float*)d_in[2];
  const float* bih_f = (const float*)d_in[3];
  const float* bhh_f = (const float*)d_in[4];
  const float* Wih_b = (const float*)d_in[5];
  const float* Whh_b = (const float*)d_in[6];
  const float* bih_b = (const float*)d_in[7];
  const float* bhh_b = (const float*)d_in[8];
  float* out = (float*)d_out;

  u16* wsu = (u16*)d_ws;
  u16* Wih = wsu;                                // 2 * NW bf16
  u16* Whh = wsu + 2 * (size_t)NW;               // 2 * NW bf16
  float* bias = (float*)(wsu + 4 * (size_t)NW);  // 1024 f32
  u16* hbuf = (u16*)(bias + 1024);               // 2*2*64*512 bf16
  unsigned* bar = (unsigned*)(hbuf + 131072);    // {count, gen}

  int prep_elems = 4 * NW + 1024 + 2;
  prep_kernel<<<(prep_elems + 255) / 256, 256, 0, stream>>>(
      Wih_f, Whh_f, bih_f, bhh_f, Wih_b, Whh_b, bih_b, bhh_b, Wih, Whh, bias, bar);
  xproj_kernel<<<dim3(1024, 2), 512, 0, stream>>>(x, Wih, bias, out);
  rnn_kernel<<<8, 256, 0, stream>>>(Whh, hbuf, out, bar);
}
